// Data2DataCrossEntropyLoss_14937896256161
// MI455X (gfx1250) — matrix operand order repaired and checked by reading, not compiled, not run on hardware
//
#include <hip/hip_runtime.h>
#include <hip/hip_bf16.h>
#include <math.h>
#include <stdint.h>

#define N_ROWS 4096
#define DIM    1024
#define NLAB   8
#define NTILE  (N_ROWS / 128)              // 32 row/col tiles
#define NBLK   (NTILE * (NTILE + 1) / 2)   // 528 triangular blocks
#define TEMP_INV (1.0f/0.75f)
#define MP_C   0.98f
#define LDSTRIDE 40   // ushorts per LDS row (64B data + 16B pad)

typedef __attribute__((ext_vector_type(16))) __bf16    v16bf;
typedef __attribute__((ext_vector_type(8)))  float     v8f;
typedef __attribute__((ext_vector_type(4)))  int       v4i;
typedef __attribute__((ext_vector_type(8)))  int       v8i;
typedef __attribute__((ext_vector_type(4)))  unsigned  v4u;
typedef __attribute__((address_space(1))) v4i gv4i;
typedef __attribute__((address_space(3))) v4i lv4i;

union FragAB { uint4 q[2]; v16bf v; };

// ---------------- CDNA5 data movers (probe-verified on this toolchain) -----
#if defined(__has_builtin)
#if __has_builtin(__builtin_amdgcn_tensor_load_to_lds) && __has_builtin(__builtin_amdgcn_s_wait_tensorcnt)
#define HAVE_TDM 1
#endif
#if __has_builtin(__builtin_amdgcn_global_load_async_to_lds_b128)
#define HAVE_ASYNC_LDS 1
#endif
#if __has_builtin(__builtin_amdgcn_s_wait_asynccnt)
#define HAVE_WAIT_ASYNC 1
#endif
#endif

__device__ __forceinline__ void async_copy_b128(void* lds_dst, const void* gsrc) {
#ifdef HAVE_ASYNC_LDS
  __builtin_amdgcn_global_load_async_to_lds_b128(
      (gv4i*)(unsigned long long)(uintptr_t)gsrc,
      (lv4i*)(unsigned)(uintptr_t)lds_dst, 0, 0);
#else
  *(uint4*)lds_dst = *(const uint4*)gsrc;
#endif
}

#ifdef HAVE_TDM
// TDM D# per CDNA5 ISA: one instruction moves a 128x32
// bf16 tile global->LDS with 16B padding every 64B row (80B LDS stride).
__device__ __forceinline__ void tdm_load_tile(unsigned lds_off, unsigned long long gaddr) {
  v4u g0; v8i g1;
  const v4i gz4 = {0, 0, 0, 0};
  const v8i gz8 = {0, 0, 0, 0, 0, 0, 0, 0};
  g0[0] = 1u;                                        // count=1, load-descriptor
  g0[1] = lds_off;                                   // lds_addr
  g0[2] = (unsigned)gaddr;                           // global_addr[31:0]
  g0[3] = (unsigned)(gaddr >> 32) | 0x80000000u;     // addr[56:32] | type=2
  g1[0] = (int)((1u << 16)    // data_size = 2 bytes
        |       (1u << 20)    // pad_enable
        |       (3u << 22)    // pad_interval: 16 DWORDs (64B)
        |       (3u << 25));  // pad_amount: 4 DWORDs (16B)
  g1[1] = (int)(1024u << 16);                        // tensor_dim0 = 1024
  g1[2] = (int)(4096u << 16);                        // tensor_dim1 = 4096
  g1[3] = (int)(32u << 16);                          // tile_dim0 = 32 (elems)
  g1[4] = 128;                                       // tile_dim1 = 128 (rows)
  g1[5] = 1024;                                      // tensor_dim0_stride
  g1[6] = 0;
  g1[7] = 0;
  __builtin_amdgcn_tensor_load_to_lds(g0, g1, gz4, gz4, gz8, 0);
}
#endif

// issue copies for one k-step (A tile rows ibase.., B tile rows jbase..)
__device__ __forceinline__ void tile_fill(unsigned short* dA, unsigned short* dB,
                                          const unsigned short* enb, int ibase,
                                          int jbase, int kk, bool diag,
                                          int t, int wid) {
#ifdef HAVE_TDM
  if (wid == 0) {
    tdm_load_tile((unsigned)(uintptr_t)dA,
                  (unsigned long long)(uintptr_t)(enb + (size_t)ibase * DIM + kk));
    if (!diag)
      tdm_load_tile((unsigned)(uintptr_t)dB,
                    (unsigned long long)(uintptr_t)(enb + (size_t)jbase * DIM + kk));
  }
#else
  const int crow = t >> 2, ccol = t & 3;                  // 2 b128 chunks each
  const int l0 = crow * LDSTRIDE + ccol * 8;
  const int l1 = (crow + 64) * LDSTRIDE + ccol * 8;
  const size_t g0 = (size_t)(ibase + crow) * DIM + kk + ccol * 8;
  const size_t g1 = (size_t)(ibase + crow + 64) * DIM + kk + ccol * 8;
  async_copy_b128(&dA[l0], enb + g0);
  async_copy_b128(&dA[l1], enb + g1);
  if (!diag) {
    const size_t h0 = (size_t)(jbase + crow) * DIM + kk + ccol * 8;
    const size_t h1 = (size_t)(jbase + crow + 64) * DIM + kk + ccol * 8;
    async_copy_b128(&dB[l0], enb + h0);
    async_copy_b128(&dB[l1], enb + h1);
  }
#endif
}

__device__ __forceinline__ void tile_wait() {
#ifdef HAVE_TDM
  // TENSORcnt is per-wave; waves that issued nothing pass instantly.
  __builtin_amdgcn_s_wait_tensorcnt(0);
#else
#ifdef HAVE_WAIT_ASYNC
  __builtin_amdgcn_s_wait_asynccnt(0);
#else
  asm volatile("s_wait_asynccnt 0x0" ::: "memory");
#endif
#endif
}

__device__ __forceinline__ unsigned bf16_rne(float f) {
  unsigned u = __float_as_uint(f);
  return (u + 0x7FFFu + ((u >> 16) & 1u)) >> 16;
}

// ---------------------------------------------------------------------------
// prep: per-row norms, pos_attr, packed label codes, bf16(en), accum init.
// ---------------------------------------------------------------------------
__global__ __launch_bounds__(256)
void prep_kernel(const float* __restrict__ embed, const float* __restrict__ proxy,
                 const int* __restrict__ label,
                 unsigned short* __restrict__ enb, float* __restrict__ pos_attr,
                 int* __restrict__ code, float* __restrict__ rowmax,
                 float* __restrict__ rowsum) {
  const int i = blockIdx.x;
  const int t = threadIdx.x;
  const float4 e4 = ((const float4*)(embed + (size_t)i * DIM))[t];
  const float4 p4 = ((const float4*)(proxy + (size_t)i * DIM))[t];
  float ee = e4.x*e4.x + e4.y*e4.y + e4.z*e4.z + e4.w*e4.w;
  float pp = p4.x*p4.x + p4.y*p4.y + p4.z*p4.z + p4.w*p4.w;
  float ep = e4.x*p4.x + e4.y*p4.y + e4.z*p4.z + e4.w*p4.w;
  for (int d = 16; d > 0; d >>= 1) {
    ee += __shfl_xor(ee, d, 32);
    pp += __shfl_xor(pp, d, 32);
    ep += __shfl_xor(ep, d, 32);
  }
  __shared__ float red[3][8];
  const int wid = t >> 5, lane = t & 31;
  if (lane == 0) { red[0][wid] = ee; red[1][wid] = pp; red[2][wid] = ep; }
  __syncthreads();
  float see = 0.f, spp = 0.f, sep = 0.f;
  for (int w = 0; w < 8; ++w) { see += red[0][w]; spp += red[1][w]; sep += red[2][w]; }
  const float ne  = fmaxf(sqrtf(see), 1e-8f);
  const float np_ = fmaxf(sqrtf(spp), 1e-8f);
  if (t == 0) {
    const float smp = sep / (ne * np_);
    pos_attr[i] = fmaxf((MP_C - smp) * TEMP_INV, 0.f);
    int c = 0;
    for (int l = 0; l < NLAB; ++l) c |= (label[i * NLAB + l] != 0) << l;
    code[i] = c;
    rowmax[i] = -1e30f;
    rowsum[i] = 0.f;
  }
  const float inv = 1.f / ne;
  const unsigned lo = bf16_rne(e4.x * inv) | (bf16_rne(e4.y * inv) << 16);
  const unsigned hi = bf16_rne(e4.z * inv) | (bf16_rne(e4.w * inv) << 16);
  ((uint2*)(enb + (size_t)i * DIM))[t] = make_uint2(lo, hi);
}

// ---------------------------------------------------------------------------
// simgemm: triangular 128x128 tiles, double-buffered TDM feed, bf16 WMMA,
// fused row (shuffle-reduced) + mirrored column (in-lane) epilogues.
// ---------------------------------------------------------------------------
__global__ __launch_bounds__(256, 1)
void simgemm_kernel(const unsigned short* __restrict__ enb,
                    const int* __restrict__ code,
                    float* __restrict__ rowmax, float* __restrict__ rowsum) {
  __shared__ __align__(16) unsigned short sA[2][128 * LDSTRIDE];
  __shared__ __align__(16) unsigned short sB[2][128 * LDSTRIDE];
  __shared__ int sCodeR[128], sCodeC[128];

  // decode triangular block index -> (by, bx), by <= bx
  int rem = blockIdx.x, by = 0;
  while (rem >= NTILE - by) { rem -= NTILE - by; ++by; }
  const int bx = by + rem;
  const int ibase = by * 128, jbase = bx * 128;
  const bool diag = (by == bx);

  const int t = threadIdx.x;
  if (t < 128) sCodeR[t] = code[ibase + t];
  else         sCodeC[t - 128] = code[jbase + t - 128];

  const int wid = t >> 5, lane = t & 31;
  const int waveRow = wid >> 2, waveCol = wid & 3;
  const int lh = lane >> 4;      // K-half of lane (ISA frag layout)
  const int nl = lane & 15;      // M (A) / N (B,C,D) index in tile

  v8f acc[4][2];
  const v8f vzero = {0.f, 0.f, 0.f, 0.f, 0.f, 0.f, 0.f, 0.f};
#pragma unroll
  for (int a = 0; a < 4; ++a)
#pragma unroll
    for (int b = 0; b < 2; ++b) acc[a][b] = vzero;

  tile_fill(sA[0], sB[0], enb, ibase, jbase, 0, diag, t, wid);

  for (int kk = 0; kk < DIM; kk += 32) {
    const int cur = (kk >> 5) & 1;
    tile_wait();
    __syncthreads();   // tile[cur] visible; all reads of tile[1-cur] retired
    if (kk + 32 < DIM)
      tile_fill(sA[1 - cur], sB[1 - cur], enb, ibase, jbase, kk + 32, diag, t, wid);

    const unsigned short* __restrict__ pA = sA[cur];
    const unsigned short* __restrict__ pB = diag ? sA[cur] : sB[cur];

    FragAB fa[4], fb[2];
#pragma unroll
    for (int tr = 0; tr < 4; ++tr) {     // A 16x32: K {8h..8h+7, 16+8h..}
      const int r0 = waveRow * 64 + tr * 16 + nl;
      fa[tr].q[0] = *(const uint4*)&pA[r0 * LDSTRIDE + lh * 8];
      fa[tr].q[1] = *(const uint4*)&pA[r0 * LDSTRIDE + 16 + lh * 8];
    }
#pragma unroll
    for (int tc = 0; tc < 2; ++tc) {     // B 32x16: K {8h..8h+7, 16+8h..}
      const int c0 = waveCol * 32 + tc * 16 + nl;
      fb[tc].q[0] = *(const uint4*)&pB[c0 * LDSTRIDE + lh * 8];
      fb[tc].q[1] = *(const uint4*)&pB[c0 * LDSTRIDE + 16 + lh * 8];
    }
#pragma unroll
    for (int tr = 0; tr < 4; ++tr)
#pragma unroll
      for (int tc = 0; tc < 2; ++tc)
        acc[tr][tc] = __builtin_amdgcn_wmma_f32_16x16x32_bf16(
            false, fa[tr].v, false, fb[tc].v, (short)0, acc[tr][tc],
            false, false);
    // no trailing barrier: next iteration's barrier orders refill vs reads
  }

  // fused epilogue — fully unrolled so acc[][][] indices are compile-time
  // constants (dynamic VGPR-tuple indexing would force scratch spills).
  const float SHIFT = MP_C - 1.0f;
  const int J0 = jbase + waveCol * 32 + nl;
  const int J1 = J0 + 16;
  const int cJ0 = sCodeC[waveCol * 32 + nl];
  const int cJ1 = sCodeC[waveCol * 32 + 16 + nl];
  float colMax0 = -1e30f, colMax1 = -1e30f, colSum0 = 0.f, colSum1 = 0.f;

#pragma unroll
  for (int tr = 0; tr < 4; ++tr) {
#pragma unroll
    for (int r = 0; r < 8; ++r) {
      const int rloc = waveRow * 64 + tr * 16 + lh * 8 + r;
      const int I = ibase + rloc;
      const int cI = sCodeR[rloc];
      float m = -1e30f, s = 0.f;
      {
        const float sim = (acc[tr][0][r] + SHIFT) * TEMP_INV;
        if (J0 != I) {
          m = fmaxf(m, sim);
          const float e = (cI != cJ0) ? __expf(fmaxf(sim, 0.f)) : 0.f;
          s += e;
          if (!diag) { colMax0 = fmaxf(colMax0, sim); colSum0 += e; }
        }
      }
      {
        const float sim = (acc[tr][1][r] + SHIFT) * TEMP_INV;
        if (J1 != I) {
          m = fmaxf(m, sim);
          const float e = (cI != cJ1) ? __expf(fmaxf(sim, 0.f)) : 0.f;
          s += e;
          if (!diag) { colMax1 = fmaxf(colMax1, sim); colSum1 += e; }
        }
      }
#pragma unroll
      for (int d = 1; d < 16; d <<= 1) {   // reduce across N-lanes (same row)
        m = fmaxf(m, __shfl_xor(m, d, 32));
        s += __shfl_xor(s, d, 32);
      }
      if (nl == 0) {
        atomicMax(&rowmax[I], m);
        atomicAdd(&rowsum[I], s);
      }
    }
  }
  if (!diag) {   // mirrored stats: column J of this tile == row J of sim
    atomicMax(&rowmax[J0], colMax0);
    atomicAdd(&rowsum[J0], colSum0);
    atomicMax(&rowmax[J1], colMax1);
    atomicAdd(&rowsum[J1], colSum1);
  }
}

// ---------------------------------------------------------------------------
// finalize: mean_i [ pos_attr + log(exp(-pos_attr) + rowsum*exp(-rowmax)) ]
// ---------------------------------------------------------------------------
__global__ __launch_bounds__(256)
void finalize_kernel(const float* __restrict__ pos_attr,
                     const float* __restrict__ rowmax,
                     const float* __restrict__ rowsum,
                     float* __restrict__ out) {
  __shared__ float red[8];
  const int t = threadIdx.x;
  float acc = 0.f;
  for (int i = t; i < N_ROWS; i += 256) {
    const float pa = pos_attr[i];
    acc += pa + __logf(__expf(-pa) + rowsum[i] * __expf(-rowmax[i]));
  }
  for (int d = 16; d > 0; d >>= 1) acc += __shfl_xor(acc, d, 32);
  if ((t & 31) == 0) red[t >> 5] = acc;
  __syncthreads();
  if (t == 0) {
    float s = 0.f;
    for (int w = 0; w < 8; ++w) s += red[w];
    out[0] = s * (1.0f / (float)N_ROWS);
  }
}

// ---------------------------------------------------------------------------
extern "C" void kernel_launch(void* const* d_in, const int* in_sizes, int n_in,
                              void* d_out, int out_size, void* d_ws, size_t ws_size,
                              hipStream_t stream) {
  (void)in_sizes; (void)n_in; (void)out_size; (void)ws_size;
  const float* embed = (const float*)d_in[0];
  const float* proxy = (const float*)d_in[1];
  const int*   label = (const int*)d_in[2];

  char* ws = (char*)d_ws;
  unsigned short* enb = (unsigned short*)ws;                      // 8 MB bf16 en
  float* pos_attr = (float*)(ws + (size_t)N_ROWS * DIM * 2);
  float* rowmax   = pos_attr + N_ROWS;
  float* rowsum   = rowmax + N_ROWS;
  int*   code     = (int*)(rowsum + N_ROWS);

  hipLaunchKernelGGL(prep_kernel, dim3(N_ROWS), dim3(256), 0, stream,
                     embed, proxy, label, enb, pos_attr, code, rowmax, rowsum);
  hipLaunchKernelGGL(simgemm_kernel, dim3(NBLK), dim3(256), 0, stream,
                     enb, code, rowmax, rowsum);
  hipLaunchKernelGGL(finalize_kernel, dim3(1), dim3(256), 0, stream,
                     pos_attr, rowmax, rowsum, (float*)d_out);
}
